// MultiHead_43782896615645
// MI455X (gfx1250) — matrix operand order repaired and checked by reading, not compiled, not run on hardware
//
#include <hip/hip_runtime.h>
#include <hip/hip_bf16.h>

typedef __attribute__((ext_vector_type(16))) _Float16 v16h;
typedef __attribute__((ext_vector_type(8)))  _Float16 v8h;
typedef __attribute__((ext_vector_type(8)))  float    v8f;
typedef __attribute__((ext_vector_type(4)))  unsigned int u32x4;
typedef __attribute__((ext_vector_type(8)))  int          i32x8;
typedef __attribute__((ext_vector_type(4)))  int          i32x4;

union V16 { v16h v; v8h h[2]; };

#define D_MODEL 1024
#define N_HEADS 16
#define D_K     64
#define BATCH   4
#define SEQ     2048
#define M_TOT   (BATCH * SEQ)   // 8192 rows
#define KSTAGE  128             // k-chunk staged to LDS per TDM transfer

static __device__ __forceinline__ v8f wmma16(v16h a, v16h b, v8f c) {
  // v_wmma_f32_16x16x32_f16: D = A(16x32 f16) * B(32x16 f16) + C(16x16 f32)
  return __builtin_amdgcn_wmma_f32_16x16x32_f16(false, a, false, b, (short)0, c, false, false);
}

// TDM: DMA a [64 rows x 128 k] f16 tile (row stride D_MODEL) from global to LDS.
// D# layout per CDNA5 ISA ch.8; 6-arg builtin on this toolchain.
static __device__ __forceinline__ void tdm_load_w_tile(unsigned lds_off,
                                                       const _Float16* gptr) {
  unsigned long long ga = (unsigned long long)(size_t)gptr;
  u32x4 g0;
  g0[0] = 1u;                                           // count=1, user descriptor
  g0[1] = lds_off;                                      // lds_addr (bytes)
  g0[2] = (unsigned)(ga & 0xFFFFFFFFu);                 // global_addr[31:0]
  g0[3] = (unsigned)((ga >> 32) & 0x1FFFFFFu) | (2u << 30); // addr[56:32] | type=2
  i32x8 g1;
  g1[0] = (1 << 16);                                    // data_size=1 -> 2 bytes
  g1[1] = (int)(((unsigned)D_MODEL & 0xFFFFu) << 16);   // tensor_dim0[15:0]
  g1[2] = (int)(((unsigned)D_MODEL & 0xFFFFu) << 16);   // tensor_dim1[15:0]
  g1[3] = (int)((unsigned)KSTAGE << 16);                // tile_dim0 = 128 (k, contig)
  g1[4] = 64;                                           // tile_dim1 = 64 rows
  g1[5] = D_MODEL;                                      // tensor_dim0_stride = 1024
  g1[6] = 0;
  g1[7] = 0;
  i32x4 z4 = {};                                        // 2D tensor: groups 2/3 unused
  i32x8 z8 = {};
  __builtin_amdgcn_tensor_load_to_lds(g0, g1, z4, z4, z8, 0);
}

// ---------------------------------------------------------------- convert
__global__ void mha_cvt_f32_to_f16(_Float16* __restrict__ dst,
                                   const float* __restrict__ src, int n) {
  int i = blockIdx.x * blockDim.x + threadIdx.x;
  int stride = gridDim.x * blockDim.x;
  for (; i < n; i += stride) dst[i] = (_Float16)src[i];
}

// ---------------------------------------------------------------- QKV projection
// Y[m, e] = sum_d X[m,d] * W[e,d]   (x @ W^T), M=8192, N=1024, K=1024
// Block: 8 waves, each a 32x64 tile sharing the same 64 weight rows (n0..n0+63).
// Weight tile double-buffered in LDS via Tensor Data Mover.
__global__ __launch_bounds__(256) void mha_gemm_qkv(
    const _Float16* __restrict__ Xh,
    const _Float16* __restrict__ Wq, const _Float16* __restrict__ Wk,
    const _Float16* __restrict__ Wv,
    _Float16* __restrict__ Qh, _Float16* __restrict__ Kh,
    _Float16* __restrict__ Vt) {
  __shared__ __align__(128) _Float16 wlds[2][64 * KSTAGE];   // 2 x 16KB

  const int wave = threadIdx.x >> 5;
  const int lane = threadIdx.x & 31;
  const int lo16 = lane & 15;
  const int hi   = lane >> 4;           // half-wave select
  const int which = blockIdx.y;         // 0=Q 1=K 2=V
  const _Float16* W = (which == 0) ? Wq : (which == 1) ? Wk : Wv;

  const int t  = blockIdx.x * 8 + wave; // 4096 wave-tiles of 32x64
  const int mt = t & 255;               // 256 M-tiles of 32
  const int nt = t >> 8;                // 16 N-tiles of 64 (uniform within block)
  const int m0 = mt * 32;
  const int n0 = nt * 64;

  v8f acc[2][4] = {};
  const _Float16* arow0 = Xh + (size_t)(m0 + lo16) * D_MODEL;
  const _Float16* arow1 = Xh + (size_t)(m0 + 16 + lo16) * D_MODEL;
  const _Float16* wbase = W + (size_t)n0 * D_MODEL;

  // prologue: stage k-chunk 0
  if (wave == 0)
    tdm_load_w_tile((unsigned)(size_t)&wlds[0][0], wbase);

  for (int stage = 0; stage < D_MODEL / KSTAGE; ++stage) {
    const int k0  = stage * KSTAGE;
    const int cur = stage & 1;
    if (wave == 0) __builtin_amdgcn_s_wait_tensorcnt(0);
    __syncthreads();                    // publish wlds[cur] to all waves
    if (wave == 0 && stage + 1 < D_MODEL / KSTAGE)
      tdm_load_w_tile((unsigned)(size_t)&wlds[cur ^ 1][0],
                      wbase + (stage + 1) * KSTAGE);

    __builtin_prefetch(arow0 + k0 + KSTAGE, 0, 1);  // global_prefetch next A chunks
    __builtin_prefetch(arow1 + k0 + KSTAGE, 0, 1);

#pragma unroll
    for (int ks = 0; ks < KSTAGE; ks += 32) {
      const int abase = k0 + ks + hi * 16;          // A f16 16x32 layout
      V16 a0, a1;
      a0.h[0] = *(const v8h*)(arow0 + abase);
      a0.h[1] = *(const v8h*)(arow0 + abase + 8);
      a1.h[0] = *(const v8h*)(arow1 + abase);
      a1.h[1] = *(const v8h*)(arow1 + abase + 8);
      // distinct B-tile registers -> one batched DS clause, WMMAs back-to-back
      const _Float16* bp = &wlds[cur][lo16 * KSTAGE + ks + hi * 16];
      V16 b0, b1, b2, b3;
      b0.h[0] = *(const v8h*)(bp);                  b0.h[1] = *(const v8h*)(bp + 8);
      b1.h[0] = *(const v8h*)(bp + 16 * KSTAGE);    b1.h[1] = *(const v8h*)(bp + 16 * KSTAGE + 8);
      b2.h[0] = *(const v8h*)(bp + 32 * KSTAGE);    b2.h[1] = *(const v8h*)(bp + 32 * KSTAGE + 8);
      b3.h[0] = *(const v8h*)(bp + 48 * KSTAGE);    b3.h[1] = *(const v8h*)(bp + 48 * KSTAGE + 8);
      acc[0][0] = wmma16(a0.v, b0.v, acc[0][0]);
      acc[0][1] = wmma16(a0.v, b1.v, acc[0][1]);
      acc[0][2] = wmma16(a0.v, b2.v, acc[0][2]);
      acc[0][3] = wmma16(a0.v, b3.v, acc[0][3]);
      acc[1][0] = wmma16(a1.v, b0.v, acc[1][0]);
      acc[1][1] = wmma16(a1.v, b1.v, acc[1][1]);
      acc[1][2] = wmma16(a1.v, b2.v, acc[1][2]);
      acc[1][3] = wmma16(a1.v, b3.v, acc[1][3]);
    }
  }

  // scatter epilogue into head layouts
#pragma unroll
  for (int mi = 0; mi < 2; ++mi) {
#pragma unroll
    for (int tn = 0; tn < 4; ++tn) {
#pragma unroll
      for (int r = 0; r < 8; ++r) {
        const int m = m0 + mi * 16 + r + 8 * hi;  // C layout: M = vgpr + 8*(lane>=16)
        const int bb = m >> 11;                   // batch
        const int s  = m & (SEQ - 1);
        const int e  = n0 + tn * 16 + lo16;
        const int h  = e >> 6;
        const int d  = e & 63;
        const size_t bh = (size_t)bb * N_HEADS + h;
        const _Float16 val = (_Float16)acc[mi][tn][r];
        if (which == 2) {
          Vt[(bh * D_K + d) * SEQ + s] = val;
        } else {
          _Float16* dst = (which == 0) ? Qh : Kh;
          dst[(bh * SEQ + s) * D_K + d] = val;
        }
      }
    }
  }
}

// ---------------------------------------------------------------- RoPE (in place, f16)
__global__ void mha_rope(_Float16* __restrict__ Qh, _Float16* __restrict__ Kh,
                         const int* __restrict__ pos) {
  const int i = blockIdx.x * blockDim.x + threadIdx.x;   // BH*S*32 pairs
  if (i >= BATCH * N_HEADS * SEQ * 32) return;
  const int p  = i & 31;
  const int s  = (i >> 5) & (SEQ - 1);
  const int bh = i >> 16;               // 32*2048 = 65536
  // inv_freq = theta^(-2p/64); log2(10000) = 13.28771237954945
  const float inv = exp2f(-(float)(2 * p) * (1.0f / 64.0f) * 13.28771237954945f);
  const float ang = (float)pos[s] * inv;
  const float c = __cosf(ang), sn = __sinf(ang);
  const size_t base = ((size_t)bh * SEQ + s) * D_K + 2 * p;
  {
    float x1 = (float)Qh[base], x2 = (float)Qh[base + 1];
    Qh[base]     = (_Float16)(x1 * c - x2 * sn);
    Qh[base + 1] = (_Float16)(x1 * sn + x2 * c);
  }
  {
    float x1 = (float)Kh[base], x2 = (float)Kh[base + 1];
    Kh[base]     = (_Float16)(x1 * c - x2 * sn);
    Kh[base + 1] = (_Float16)(x1 * sn + x2 * c);
  }
}

// ---------------------------------------------------------------- flash attention
// One wave owns a 16-query tile of one (b,h); loops causal key blocks of 32.
__global__ __launch_bounds__(256) void mha_flash(
    const _Float16* __restrict__ Qh, const _Float16* __restrict__ Kh,
    const _Float16* __restrict__ Vt, _Float16* __restrict__ Oh) {
  __shared__ __align__(32) _Float16 plds[8][16 * 32];   // per-wave P staging

  const int wave = threadIdx.x >> 5;
  const int lane = threadIdx.x & 31;
  const int lo16 = lane & 15;
  const int hi   = lane >> 4;

  const int qchunk = blockIdx.x & 15;   // 16 chunks of 128 queries
  const int bh     = blockIdx.x >> 4;   // 0..63
  const int q0     = (qchunk * 8 + wave) * 16;

  const _Float16* Qb = Qh + (size_t)bh * SEQ * D_K;
  const _Float16* Kb = Kh + (size_t)bh * SEQ * D_K;
  const _Float16* Vb = Vt + (size_t)bh * D_K * SEQ;

  // Q tile 16x64 as two A-chunks (d 0-31, 32-63), resident for the whole loop
  V16 aQ[2];
  {
    const _Float16* qp = Qb + (size_t)(q0 + lo16) * D_K;
    const int base = hi * 16;
    aQ[0].h[0] = *(const v8h*)(qp + base);
    aQ[0].h[1] = *(const v8h*)(qp + base + 8);
    aQ[1].h[0] = *(const v8h*)(qp + 32 + base);
    aQ[1].h[1] = *(const v8h*)(qp + 32 + base + 8);
  }

  v8f oacc[4] = {};
  float mrow[8], lrow[8];
#pragma unroll
  for (int r = 0; r < 8; ++r) { mrow[r] = -3.0e38f; lrow[r] = 0.0f; }

  const float LOG2E = 1.44269504088896340736f;
  const float SC    = 0.125f;           // 1/sqrt(64)
  const int kend = q0 + 16;             // causal: keys < q0+16

  for (int kb = 0; kb < kend; kb += 32) {
    // ---- scores: two 16x16 C-tiles; batch all 4 B loads, then 4 WMMAs
    const _Float16* kp0 = Kb + (size_t)(kb + lo16) * D_K + hi * 16;
    const _Float16* kp1 = Kb + (size_t)(kb + 16 + lo16) * D_K + hi * 16;
    V16 bk00, bk01, bk10, bk11;         // [keytile][dchunk]
    bk00.h[0] = *(const v8h*)(kp0);       bk00.h[1] = *(const v8h*)(kp0 + 8);
    bk01.h[0] = *(const v8h*)(kp0 + 32);  bk01.h[1] = *(const v8h*)(kp0 + 40);
    bk10.h[0] = *(const v8h*)(kp1);       bk10.h[1] = *(const v8h*)(kp1 + 8);
    bk11.h[0] = *(const v8h*)(kp1 + 32);  bk11.h[1] = *(const v8h*)(kp1 + 40);
    v8f s0 = {}, s1 = {};
    s0 = wmma16(aQ[0].v, bk00.v, s0);
    s1 = wmma16(aQ[0].v, bk10.v, s1);
    s0 = wmma16(aQ[1].v, bk01.v, s0);
    s1 = wmma16(aQ[1].v, bk11.v, s1);

    // ---- online softmax (row = vgpr r + 8*hi, lives across 16 lanes)
    float p0[8], p1[8], rm[8];
    const bool needmask = (kb + 31 > q0);
#pragma unroll
    for (int r = 0; r < 8; ++r) {
      float v0 = s0[r] * SC, v1 = s1[r] * SC;
      if (needmask) {
        const int row = q0 + r + 8 * hi;
        if (kb + lo16 > row)      v0 = -3.0e38f;
        if (kb + 16 + lo16 > row) v1 = -3.0e38f;
      }
      p0[r] = v0; p1[r] = v1;
      rm[r] = fmaxf(v0, v1);
    }
#pragma unroll
    for (int off = 1; off < 16; off <<= 1)   // reduce within each 16-lane half
#pragma unroll
      for (int r = 0; r < 8; ++r) rm[r] = fmaxf(rm[r], __shfl_xor(rm[r], off));

    float alpha[8], rs[8];
#pragma unroll
    for (int r = 0; r < 8; ++r) {
      const float mn = fmaxf(mrow[r], rm[r]);
      alpha[r] = exp2f((mrow[r] - mn) * LOG2E);
      mrow[r]  = mn;
      p0[r] = exp2f((p0[r] - mn) * LOG2E);
      p1[r] = exp2f((p1[r] - mn) * LOG2E);
      rs[r] = p0[r] + p1[r];
    }
#pragma unroll
    for (int off = 1; off < 16; off <<= 1)
#pragma unroll
      for (int r = 0; r < 8; ++r) rs[r] += __shfl_xor(rs[r], off);
#pragma unroll
    for (int r = 0; r < 8; ++r) {
      lrow[r] = lrow[r] * alpha[r] + rs[r];
#pragma unroll
      for (int t = 0; t < 4; ++t) oacc[t][r] *= alpha[r];
    }

    // ---- P: C-layout -> LDS (row major 16x32) -> reload in A-layout
#pragma unroll
    for (int r = 0; r < 8; ++r) {
      const int m = r + 8 * hi;
      plds[wave][m * 32 + lo16]      = (_Float16)p0[r];
      plds[wave][m * 32 + 16 + lo16] = (_Float16)p1[r];
    }
    asm volatile("s_wait_dscnt 0" ::: "memory");   // per-wave LDS store->load hazard

    V16 aP;
    {
      const _Float16* pp = &plds[wave][lo16 * 32 + hi * 16];
      aP.h[0] = *(const v8h*)(pp);
      aP.h[1] = *(const v8h*)(pp + 8);
    }

    // ---- PV: batch the 4 V B-tiles (transposed V -> row loads), then 4 WMMAs
    const _Float16* vp = Vb + (size_t)lo16 * SEQ + kb + hi * 16;
    V16 bv0, bv1, bv2, bv3;
    bv0.h[0] = *(const v8h*)(vp);                 bv0.h[1] = *(const v8h*)(vp + 8);
    bv1.h[0] = *(const v8h*)(vp + 16 * SEQ);      bv1.h[1] = *(const v8h*)(vp + 16 * SEQ + 8);
    bv2.h[0] = *(const v8h*)(vp + 32 * SEQ);      bv2.h[1] = *(const v8h*)(vp + 32 * SEQ + 8);
    bv3.h[0] = *(const v8h*)(vp + 48 * SEQ);      bv3.h[1] = *(const v8h*)(vp + 48 * SEQ + 8);
    oacc[0] = wmma16(aP.v, bv0.v, oacc[0]);
    oacc[1] = wmma16(aP.v, bv1.v, oacc[1]);
    oacc[2] = wmma16(aP.v, bv2.v, oacc[2]);
    oacc[3] = wmma16(aP.v, bv3.v, oacc[3]);
  }

  // ---- normalize + write O as f16 in [B, S, E] layout for the final GEMM
  const int bb = bh >> 4, h = bh & 15;
#pragma unroll
  for (int t = 0; t < 4; ++t) {
#pragma unroll
    for (int r = 0; r < 8; ++r) {
      const int row = q0 + r + 8 * hi;
      const int e   = h * D_K + t * 16 + lo16;
      Oh[((size_t)bb * SEQ + row) * D_MODEL + e] = (_Float16)(oacc[t][r] / lrow[r]);
    }
  }
}

// ---------------------------------------------------------------- output projection
// out[m, n] = sum_e O[m,e] * WO[n,e]  -> f32 to d_out.  Same TDM/LDS scheme.
__global__ __launch_bounds__(256) void mha_gemm_out(
    const _Float16* __restrict__ Oh, const _Float16* __restrict__ Wo,
    float* __restrict__ out) {
  __shared__ __align__(128) _Float16 wlds[2][64 * KSTAGE];

  const int wave = threadIdx.x >> 5;
  const int lane = threadIdx.x & 31;
  const int lo16 = lane & 15;
  const int hi   = lane >> 4;

  const int t  = blockIdx.x * 8 + wave; // 4096 wave-tiles of 32x64
  const int mt = t & 255;
  const int nt = t >> 8;
  const int m0 = mt * 32;
  const int n0 = nt * 64;

  v8f acc[2][4] = {};
  const _Float16* arow0 = Oh + (size_t)(m0 + lo16) * D_MODEL;
  const _Float16* arow1 = Oh + (size_t)(m0 + 16 + lo16) * D_MODEL;
  const _Float16* wbase = Wo + (size_t)n0 * D_MODEL;

  if (wave == 0)
    tdm_load_w_tile((unsigned)(size_t)&wlds[0][0], wbase);

  for (int stage = 0; stage < D_MODEL / KSTAGE; ++stage) {
    const int k0  = stage * KSTAGE;
    const int cur = stage & 1;
    if (wave == 0) __builtin_amdgcn_s_wait_tensorcnt(0);
    __syncthreads();
    if (wave == 0 && stage + 1 < D_MODEL / KSTAGE)
      tdm_load_w_tile((unsigned)(size_t)&wlds[cur ^ 1][0],
                      wbase + (stage + 1) * KSTAGE);

    __builtin_prefetch(arow0 + k0 + KSTAGE, 0, 1);
    __builtin_prefetch(arow1 + k0 + KSTAGE, 0, 1);

#pragma unroll
    for (int ks = 0; ks < KSTAGE; ks += 32) {
      const int abase = k0 + ks + hi * 16;
      V16 a0, a1;
      a0.h[0] = *(const v8h*)(arow0 + abase);
      a0.h[1] = *(const v8h*)(arow0 + abase + 8);
      a1.h[0] = *(const v8h*)(arow1 + abase);
      a1.h[1] = *(const v8h*)(arow1 + abase + 8);
      const _Float16* bp = &wlds[cur][lo16 * KSTAGE + ks + hi * 16];
      V16 b0, b1, b2, b3;
      b0.h[0] = *(const v8h*)(bp);                  b0.h[1] = *(const v8h*)(bp + 8);
      b1.h[0] = *(const v8h*)(bp + 16 * KSTAGE);    b1.h[1] = *(const v8h*)(bp + 16 * KSTAGE + 8);
      b2.h[0] = *(const v8h*)(bp + 32 * KSTAGE);    b2.h[1] = *(const v8h*)(bp + 32 * KSTAGE + 8);
      b3.h[0] = *(const v8h*)(bp + 48 * KSTAGE);    b3.h[1] = *(const v8h*)(bp + 48 * KSTAGE + 8);
      acc[0][0] = wmma16(a0.v, b0.v, acc[0][0]);
      acc[0][1] = wmma16(a0.v, b1.v, acc[0][1]);
      acc[0][2] = wmma16(a0.v, b2.v, acc[0][2]);
      acc[0][3] = wmma16(a0.v, b3.v, acc[0][3]);
      acc[1][0] = wmma16(a1.v, b0.v, acc[1][0]);
      acc[1][1] = wmma16(a1.v, b1.v, acc[1][1]);
      acc[1][2] = wmma16(a1.v, b2.v, acc[1][2]);
      acc[1][3] = wmma16(a1.v, b3.v, acc[1][3]);
    }
  }

#pragma unroll
  for (int mi = 0; mi < 2; ++mi) {
#pragma unroll
    for (int tn = 0; tn < 4; ++tn) {
#pragma unroll
      for (int r = 0; r < 8; ++r) {
        const int m = m0 + mi * 16 + r + 8 * hi;
        const int n = n0 + tn * 16 + lo16;
        out[(size_t)m * D_MODEL + n] = acc[mi][tn][r];
      }
    }
  }
}

// ---------------------------------------------------------------- launch
extern "C" void kernel_launch(void* const* d_in, const int* in_sizes, int n_in,
                              void* d_out, int out_size, void* d_ws, size_t ws_size,
                              hipStream_t stream) {
  const float* x   = (const float*)d_in[0];
  const int*   pos = (const int*)d_in[1];
  const float* WQ  = (const float*)d_in[2];
  const float* WK  = (const float*)d_in[3];
  const float* WV  = (const float*)d_in[4];
  const float* WO  = (const float*)d_in[5];
  float* out = (float*)d_out;

  char* w = (char*)d_ws;
  const size_t X_ELEMS = (size_t)M_TOT * D_MODEL;             // 8388608
  const size_t W_ELEMS = (size_t)D_MODEL * D_MODEL;           // 1048576
  const size_t H_ELEMS = (size_t)BATCH * N_HEADS * SEQ * D_K; // 8388608

  _Float16* xh  = (_Float16*)w; w += X_ELEMS * 2;
  _Float16* wqh = (_Float16*)w; w += W_ELEMS * 2;
  _Float16* wkh = (_Float16*)w; w += W_ELEMS * 2;
  _Float16* wvh = (_Float16*)w; w += W_ELEMS * 2;
  _Float16* woh = (_Float16*)w; w += W_ELEMS * 2;
  _Float16* Qh  = (_Float16*)w; w += H_ELEMS * 2;
  _Float16* Kh  = (_Float16*)w; w += H_ELEMS * 2;
  _Float16* Vt  = (_Float16*)w; w += H_ELEMS * 2;
  _Float16* Oh  = (_Float16*)w; w += X_ELEMS * 2;

  // 1) downconvert
  mha_cvt_f32_to_f16<<<4096, 256, 0, stream>>>(xh, x, (int)X_ELEMS);
  mha_cvt_f32_to_f16<<<2048, 256, 0, stream>>>(wqh, WQ, (int)W_ELEMS);
  mha_cvt_f32_to_f16<<<2048, 256, 0, stream>>>(wkh, WK, (int)W_ELEMS);
  mha_cvt_f32_to_f16<<<2048, 256, 0, stream>>>(wvh, WV, (int)W_ELEMS);
  mha_cvt_f32_to_f16<<<2048, 256, 0, stream>>>(woh, WO, (int)W_ELEMS);

  // 2) Q/K/V projections (4096 wave-tiles of 32x64 -> 512 blocks, y = matrix)
  mha_gemm_qkv<<<dim3(512, 3), 256, 0, stream>>>(xh, wqh, wkh, wvh, Qh, Kh, Vt);

  // 3) RoPE on Q,K (BH*S*32 pairs)
  mha_rope<<<16384, 256, 0, stream>>>(Qh, Kh, pos);

  // 4) flash attention: 64 (b,h) x 16 query-chunks of 128
  mha_flash<<<1024, 256, 0, stream>>>(Qh, Kh, Vt, Oh);

  // 5) output projection -> f32
  mha_gemm_out<<<512, 256, 0, stream>>>(Oh, woh, out);
}
